// RNNCute_51316269253339
// MI455X (gfx1250) — matrix operand order repaired and checked by reading, not compiled, not run on hardware
//
#include <hip/hip_runtime.h>
#include <hip/hip_bf16.h>

typedef __attribute__((ext_vector_type(16))) __bf16 v16bf;
typedef __attribute__((ext_vector_type(8)))  float  v8f;
typedef __attribute__((ext_vector_type(4)))  unsigned int u32x4;
typedef __attribute__((ext_vector_type(8)))  int  i32x8;
typedef __attribute__((ext_vector_type(4)))  int  i32x4;

constexpr int Bdim = 4, Sdim = 4096, INdim = 1024, Hh = 16, Dd = 64, OUTdim = 1024;
constexpr int HD = Hh * Dd;        // 1024
constexpr int M1 = Bdim * Sdim;    // 16384

// ---------------------------------------------------------------------------
// fp32 -> bf16 weight conversion
// ---------------------------------------------------------------------------
__global__ void cvt_f32_bf16(const float* __restrict__ src,
                             __bf16* __restrict__ dst, int n) {
    int i = blockIdx.x * blockDim.x + threadIdx.x;
    if (i < n) dst[i] = (__bf16)src[i];
}

// ---------------------------------------------------------------------------
// TDM: issue a 2D bf16 tile load (tile_k x tile_rows) from a [rows][1024]
// row-major bf16 tensor into LDS, with HW padding of 16B after every 64B
// (-> 80B LDS row pitch, matching the conflict-free fragment layout).
// Issued by a single wave; tracked by TENSORcnt.
// ---------------------------------------------------------------------------
__device__ __forceinline__ void tdm_load_tile_bf16(
        unsigned lds_off, const __bf16* gtile,
        unsigned tile_k, unsigned tile_rows) {
    unsigned long long ga = (unsigned long long)(uintptr_t)gtile;
    u32x4 g0;
    g0[0] = 1u;                                   // count=1, user descriptor
    g0[1] = lds_off;                              // LDS byte address
    g0[2] = (unsigned)(ga & 0xffffffffu);         // global_addr[31:0]
    g0[3] = (unsigned)((ga >> 32) & 0x01ffffffu)  // global_addr[56:32]
          | (2u << 30);                           // type = 2 (image)
    i32x8 g1;
    g1[0] = (int)((1u << 16)      // data_size = 2 bytes
                | (1u << 20)      // pad_enable
                | (3u << 22)      // pad_interval: 16 dwords = 64B
                | (3u << 25));    // pad_amount:    4 dwords = 16B
    g1[1] = (int)(1024u << 16);   // tensor_dim0[15:0] = 1024 (K extent)
    g1[2] = (int)(1024u << 16);   // dim0[31:16]=0 | tensor_dim1[15:0] = 1024
    g1[3] = (int)(tile_k << 16);  // dim1[31:16]=0 | tile_dim0
    g1[4] = (int)tile_rows;       // tile_dim1 | tile_dim2=0
    g1[5] = 1024;                 // tensor_dim0_stride[31:0] = 1024 elems
    g1[6] = 0;                    // stride[47:32]=0 | dim1_stride[15:0]=0
    g1[7] = 0;
    i32x4 z4 = {0, 0, 0, 0};
#if __has_builtin(__builtin_amdgcn_tensor_load_to_lds)
#if __clang_major__ >= 23
    i32x8 z8 = {0, 0, 0, 0, 0, 0, 0, 0};
    __builtin_amdgcn_tensor_load_to_lds(g0, g1, z4, z4, z8, 0);
#else
    __builtin_amdgcn_tensor_load_to_lds(g0, g1, z4, z4, 0);
#endif
#else
    (void)g0; (void)g1; (void)z4;
#endif
}

__device__ __forceinline__ void wait_tensorcnt0() {
#if __has_builtin(__builtin_amdgcn_s_wait_tensorcnt)
    __builtin_amdgcn_s_wait_tensorcnt(0);
#else
    asm volatile("s_wait_tensorcnt 0x0" ::: "memory");
#endif
}

// Async bf16 16B copy global->LDS (per lane), tracked by ASYNCcnt.
// Builtin expects typed addrspace pointers: (as1 i32x4*, as3 i32x4*, ...)
typedef __attribute__((address_space(1))) i32x4 g_i32x4;
typedef __attribute__((address_space(3))) i32x4 l_i32x4;

__device__ __forceinline__ void async_load_b128(unsigned lds_off,
                                                const __bf16* g) {
    unsigned long long ga = (unsigned long long)(uintptr_t)g;
#if __has_builtin(__builtin_amdgcn_global_load_async_to_lds_b128)
    __builtin_amdgcn_global_load_async_to_lds_b128(
        (g_i32x4*)(uintptr_t)ga,
        (l_i32x4*)(uintptr_t)lds_off, 0, 0);
#else
    asm volatile("global_load_async_to_lds_b128 %0, %1, off"
                 :: "v"(lds_off), "v"(ga) : "memory");
#endif
}

__device__ __forceinline__ void wait_asynccnt0() {
#if __has_builtin(__builtin_amdgcn_s_wait_asynccnt)
    __builtin_amdgcn_s_wait_asynccnt(0);
#else
    asm volatile("s_wait_asynccnt 0x0" ::: "memory");
#endif
}

// ---------------------------------------------------------------------------
// bf16 WMMA GEMM: C[M=16384][N=1024] = A[M][K=1024] * B[N][K]^T
// WG tile 128x128, K-step 32, double-buffered LDS.
// B tiles: TDM DMA. A tiles: async-to-LDS (bf16) or VALU-convert (fp32).
// 8 waves as 4(M) x 2(N); each wave owns 2x4 tiles of 16x16.
// ---------------------------------------------------------------------------
template <int A_IS_F32>
__global__ __launch_bounds__(256) void gemm_bf16_wmma(
        const void* __restrict__ Aptr,
        const __bf16* __restrict__ Bw,
        float* __restrict__ C) {
    __shared__ __bf16 Alds[2][128][40];   // 80B row pitch (20 words): no conflicts
    __shared__ __bf16 Blds[2][128][40];

    const int tid  = threadIdx.x;
    const int lane = tid & 31;
    const int wid  = tid >> 5;
    const int wm   = wid >> 1;   // 0..3
    const int wn   = wid & 1;    // 0..1
    const int lr   = lane & 15;
    const int hi   = lane >> 4;

    const int mWG = blockIdx.y * 128;
    const int nWG = blockIdx.x * 128;

    v8f acc[2][4];
    #pragma unroll
    for (int mt = 0; mt < 2; ++mt)
        #pragma unroll
        for (int nt = 0; nt < 4; ++nt)
            #pragma unroll
            for (int r = 0; r < 8; ++r)
                acc[mt][nt][r] = 0.0f;

    // staging map: thread t -> row (t>>1), K segment ((t&1)*16)
    const int sr = tid >> 1;
    const int sk = (tid & 1) * 16;

    const float*  Af = (const float*)Aptr;
    const __bf16* Ab = (const __bf16*)Aptr;

    const unsigned aldsOff[2] = { (unsigned)(uintptr_t)&Alds[0][0][0],
                                  (unsigned)(uintptr_t)&Alds[1][0][0] };
    const unsigned bldsOff[2] = { (unsigned)(uintptr_t)&Blds[0][0][0],
                                  (unsigned)(uintptr_t)&Blds[1][0][0] };
    const unsigned rowByte = (unsigned)sr * 80u + (unsigned)sk * 2u;

    // ---------------- prologue: stage tile 0 into buffer 0 ----------------
    {
        if (tid == 0) tdm_load_tile_bf16(bldsOff[0], Bw + (size_t)nWG * 1024, 32, 128);
        if constexpr (A_IS_F32) {
            const float* g = Af + (size_t)(mWG + sr) * 1024 + sk;
            float fv[16];
            #pragma unroll
            for (int j = 0; j < 16; ++j) fv[j] = g[j];
            __align__(16) __bf16 ab[16];
            #pragma unroll
            for (int j = 0; j < 16; ++j) ab[j] = (__bf16)fv[j];
            *(uint4*)&Alds[0][sr][sk]     = ((uint4*)ab)[0];
            *(uint4*)&Alds[0][sr][sk + 8] = ((uint4*)ab)[1];
        } else {
            const __bf16* g = Ab + (size_t)(mWG + sr) * 1024 + sk;
            async_load_b128(aldsOff[0] + rowByte,      g);
            async_load_b128(aldsOff[0] + rowByte + 16, g + 8);
            wait_asynccnt0();
        }
        if (tid < 32) wait_tensorcnt0();
        __syncthreads();
    }

    for (int kt = 0; kt < 1024 / 32; ++kt) {
        const int cur = kt & 1;
        const int nxt = cur ^ 1;
        const bool has_next = (kt + 1) < 32;
        const int k1 = (kt + 1) * 32;

        // ---- issue staging for tile kt+1 (overlaps with compute below) ----
        float fv[16];
        if (has_next) {
            if (tid == 0)
                tdm_load_tile_bf16(bldsOff[nxt], Bw + (size_t)nWG * 1024 + k1, 32, 128);
            if constexpr (A_IS_F32) {
                const float* g = Af + (size_t)(mWG + sr) * 1024 + k1 + sk;
                __builtin_prefetch(g + 32, 0, 1);
                #pragma unroll
                for (int j = 0; j < 16; ++j) fv[j] = g[j];
            } else {
                const __bf16* g = Ab + (size_t)(mWG + sr) * 1024 + k1 + sk;
                async_load_b128(aldsOff[nxt] + rowByte,      g);
                async_load_b128(aldsOff[nxt] + rowByte + 16, g + 8);
            }
        }

        // ---- compute on tile kt ----
        // A frag (16-bit 16x32): lanes 0-15 hold K 0..7|16..23; 16-31: 8..15|24..31
        v16bf afr[2];
        #pragma unroll
        for (int mt = 0; mt < 2; ++mt) {
            const int ml = wm * 32 + mt * 16 + lr;
            ((uint4*)&afr[mt])[0] = *(const uint4*)&Alds[cur][ml][hi * 8];
            ((uint4*)&afr[mt])[1] = *(const uint4*)&Alds[cur][ml][hi * 8 + 16];
        }
        // B frag (32x16): lanes 0-15 N=lr hold K 0..7|16..23; lanes 16-31 hold K 8..15|24..31
        v16bf bfr[4];
        #pragma unroll
        for (int nt = 0; nt < 4; ++nt) {
            const int nl = wn * 64 + nt * 16 + lr;
            ((uint4*)&bfr[nt])[0] = *(const uint4*)&Blds[cur][nl][hi * 8];
            ((uint4*)&bfr[nt])[1] = *(const uint4*)&Blds[cur][nl][hi * 8 + 16];
        }
        #pragma unroll
        for (int mt = 0; mt < 2; ++mt)
            #pragma unroll
            for (int nt = 0; nt < 4; ++nt)
                acc[mt][nt] = __builtin_amdgcn_wmma_f32_16x16x32_bf16(
                    false, afr[mt], false, bfr[nt],
                    (short)0, acc[mt][nt], false, false);

        // ---- finish staging of tile kt+1 ----
        if (has_next) {
            if constexpr (A_IS_F32) {
                __align__(16) __bf16 ab[16];
                #pragma unroll
                for (int j = 0; j < 16; ++j) ab[j] = (__bf16)fv[j];
                *(uint4*)&Alds[nxt][sr][sk]     = ((uint4*)ab)[0];
                *(uint4*)&Alds[nxt][sr][sk + 8] = ((uint4*)ab)[1];
            } else {
                wait_asynccnt0();
            }
            if (tid < 32) wait_tensorcnt0();
        }
        __syncthreads();
    }

    // ---- epilogue: C/D layout -> VGPR r holds M = r + 8*hi, N = lr ----
    #pragma unroll
    for (int mt = 0; mt < 2; ++mt)
        #pragma unroll
        for (int nt = 0; nt < 4; ++nt) {
            const int gn = nWG + wn * 64 + nt * 16 + lr;
            #pragma unroll
            for (int r = 0; r < 8; ++r) {
                const int gm = mWG + wm * 32 + mt * 16 + hi * 8 + r;
                C[(size_t)gm * 1024 + gn] = acc[mt][nt][r];
            }
        }
}

// ---------------------------------------------------------------------------
// Sequential recurrence: one workgroup per (b,h); 128 threads.
// h_t = tanh(W[h] @ h_{t-1} + bias + x_t). W rows in registers (2 lanes per
// output, 32 MACs each, ds_swizzle XOR-1 reduce), x prefetched 8 steps ahead,
// h double-buffered in LDS, one barrier per step.
// ---------------------------------------------------------------------------
__global__ __launch_bounds__(128) void rnn_scan(
        const float* __restrict__ xp,     // [B*S][1024] fp32
        const float* __restrict__ h0,     // [B][H][64]
        const float* __restrict__ Wst,    // [H][64][64]
        const float* __restrict__ bias,   // [H][64]
        __bf16* __restrict__ y) {         // [B*S][1024] bf16
    __shared__ float hbuf[2][64];

    const int bh   = blockIdx.x;
    const int b    = bh >> 4;
    const int h    = bh & 15;
    const int tid  = threadIdx.x;
    const int e    = tid >> 1;
    const int half = tid & 1;

    float w[32];
    const float* wr = Wst + ((size_t)(h * 64 + e)) * 64 + half * 32;
    #pragma unroll
    for (int j = 0; j < 32; ++j) w[j] = wr[j];
    const float bb = bias[h * 64 + e];

    if (half == 0) hbuf[0][e] = h0[(b * Hh + h) * 64 + e];
    __syncthreads();

    const float* xptr = xp + ((size_t)b * Sdim) * 1024 + h * 64 + e;
    __bf16*      yptr = y  + ((size_t)b * Sdim) * 1024 + h * 64 + e;

    float xreg[8];
    #pragma unroll
    for (int i = 0; i < 8; ++i) xreg[i] = xptr[(size_t)i * 1024];

    for (int t0 = 0; t0 < Sdim; t0 += 8) {
        #pragma unroll
        for (int u = 0; u < 8; ++u) {
            const int t = t0 + u;
            const int p = t & 1;
            float s = 0.0f;
            #pragma unroll
            for (int j = 0; j < 32; ++j)
                s += w[j] * hbuf[p][half * 32 + j];
            s += __int_as_float(
                __builtin_amdgcn_ds_swizzle(__float_as_int(s), 0x041F));
            const float pre = s + bb + xreg[u];
            if (t + 8 < Sdim) xreg[u] = xptr[(size_t)(t + 8) * 1024];
            const float e2 = __builtin_amdgcn_exp2f(pre * 2.885390081777927f);
            const float th = 1.0f - 2.0f * __builtin_amdgcn_rcpf(e2 + 1.0f);
            if (half == 0) {
                hbuf[p ^ 1][e] = th;
                yptr[(size_t)t * 1024] = (__bf16)th;
            }
            __syncthreads();
        }
    }
}

// ---------------------------------------------------------------------------
extern "C" void kernel_launch(void* const* d_in, const int* in_sizes, int n_in,
                              void* d_out, int out_size, void* d_ws, size_t ws_size,
                              hipStream_t stream) {
    const float* x    = (const float*)d_in[0];  // [B,S,IN]
    const float* h0   = (const float*)d_in[1];  // [B,H,D]
    const float* w_in = (const float*)d_in[2];  // [HD,IN]
    const float* sw   = (const float*)d_in[3];  // [H,D,D]
    const float* sb   = (const float*)d_in[4];  // [H,D]
    const float* w_o  = (const float*)d_in[5];  // [OUT,HD]
    float* out = (float*)d_out;                 // [B,S,OUT]

    char* wsb = (char*)d_ws;
    size_t off = 0;
    float* xp = (float*)(wsb + off);    off += (size_t)M1 * HD * sizeof(float); // 64MB
    __bf16* ybf = (__bf16*)(wsb + off); off += (size_t)M1 * HD * 2;             // 32MB
    __bf16* wib = (__bf16*)(wsb + off); off += (size_t)HD * INdim * 2;          // 2MB
    __bf16* wob = (__bf16*)(wsb + off);                                         // 2MB

    const int nw = HD * INdim;
    cvt_f32_bf16<<<(nw + 255) / 256, 256, 0, stream>>>(w_in, wib, nw);
    cvt_f32_bf16<<<(nw + 255) / 256, 256, 0, stream>>>(w_o, wob, nw);

    // xp = x @ w_in^T
    gemm_bf16_wmma<1><<<dim3(HD / 128, M1 / 128), 256, 0, stream>>>(x, wib, xp);
    // sequential tanh recurrence, y in bf16
    rnn_scan<<<Bdim * Hh, 128, 0, stream>>>(xp, h0, sw, sb, ybf);
    // out = y @ w_out^T
    gemm_bf16_wmma<0><<<dim3(OUTdim / 128, M1 / 128), 256, 0, stream>>>(ybf, wob, out);
}
